// Model_18717467476397
// MI455X (gfx1250) — matrix operand order repaired and checked by reading, not compiled, not run on hardware
//
#include <hip/hip_runtime.h>
#include <hip/hip_bf16.h>

// ---------------- problem constants (compile-time) ----------------
#define EMB      64
#define TD       128          // TREE_DIM
#define IND      192          // 3*EMB
#define DEPTH_   8
#define BATCH_   256
#define NLEAF    65536        // BATCH << DEPTH
#define NNODES   130816       // BATCH * ((1<<(DEPTH+1))-1)
#define KCAT     320          // IND + TD (fused [W;U] GEMM K)

typedef __attribute__((ext_vector_type(16))) __bf16 v16bf;
typedef __attribute__((ext_vector_type(8)))  __bf16 bf16x8;
typedef __attribute__((ext_vector_type(8)))  float  v8f;

__device__ __forceinline__ float sigf(float x) { return 1.0f / (1.0f + __expf(-x)); }

// ---------------- weight prep: WT[n*(kW+kU) + k] = (k<kW ? W[k][n] : U[k-kW][n]) in bf16
__global__ void k_build_wt(const float* __restrict__ W, const float* __restrict__ U,
                           __bf16* __restrict__ WT, int nOut, int kW, int kU)
{
    int idx = blockIdx.x * blockDim.x + threadIdx.x;
    int ldk = kW + kU;
    if (idx >= nOut * ldk) return;
    int n = idx / ldk, k = idx % ldk;
    float v = (k < kW) ? W[k * nOut + n] : U[(k - kW) * nOut + n];
    WT[idx] = (__bf16)v;
}

// ---------------- feats = concat(emb[tok], sort[srt], const) in bf16, N x 192
__global__ void k_feats(const int* __restrict__ tok, const int* __restrict__ srt,
                        const float* __restrict__ cst, const float* __restrict__ emb,
                        const float* __restrict__ sortt, __bf16* __restrict__ feats)
{
    int idx = blockIdx.x * blockDim.x + threadIdx.x;
    if (idx >= NNODES * IND) return;
    int row = idx / IND, col = idx % IND;
    float v;
    if (col < EMB)          v = emb[(size_t)tok[row] * EMB + col];
    else if (col < 2 * EMB) v = sortt[(size_t)srt[row] * EMB + (col - EMB)];
    else                    v = cst[(size_t)row * EMB + (col - 2 * EMB)];
    feats[idx] = (__bf16)v;
}

// ---------------- generic WMMA bf16 GEMM: C[MxN] = A[MxK] * Bt[NxK]^T + bias[N]
// block = 256 threads = 8 waves; wave -> 32x64 output tile (2 M sub-tiles share B);
// B panel (64 x K, <= 40KB) staged once in LDS, reused by all 8 waves.
// grid = (N/64, M/256). M % 256 == 0, N % 64 == 0, K % 32 == 0 guaranteed by caller.
__global__ void __launch_bounds__(256) k_gemm(const __bf16* __restrict__ A, int lda,
                       const __bf16* __restrict__ Bt, int ldb,
                       const float* __restrict__ bias,
                       float* __restrict__ C, int ldc, int K)
{
    __shared__ __bf16 ldsB[64 * (KCAT + 8)];   // +8 element row pad -> conflict-free b128 reads
    const int ldsK = K + 8;
    int tid  = threadIdx.x;
    int lane = tid & 31;
    int wv   = tid >> 5;
    int half = lane >> 4;      // K-half select (ISA 16-bit A/B layouts)
    int r    = lane & 15;      // row (A) / col (B,C) within tile
    int m0 = blockIdx.y * 256 + wv * 32;
    int n0 = blockIdx.x * 64;

    // cooperative fill of the 64 x K B panel into LDS (16B chunks)
    int chunksPerRow = K >> 3;
    for (int i = tid; i < 64 * chunksPerRow; i += 256) {
        int row = i / chunksPerRow;
        int ch  = i - row * chunksPerRow;
        *(bf16x8*)(ldsB + row * ldsK + ch * 8) =
            *(const bf16x8*)(Bt + (size_t)(n0 + row) * ldb + ch * 8);
    }
    __syncthreads();

    v8f acc[2][4] = {};
    const __bf16* arow0 = A + (size_t)(m0 + r) * lda;
    const __bf16* arow1 = A + (size_t)(m0 + 16 + r) * lda;

    for (int k0 = 0; k0 < K; k0 += 32) {
        // A fragments: lanes 0-15 K = k0+0..7 & k0+16..23 ; lanes 16-31 +8
        v16bf a0, a1;
        {
            bf16x8 lo = *(const bf16x8*)(arow0 + k0 + 8 * half);
            bf16x8 hi = *(const bf16x8*)(arow0 + k0 + 16 + 8 * half);
            #pragma unroll
            for (int i = 0; i < 8; i++) { a0[i] = lo[i]; a0[8 + i] = hi[i]; }
        }
        {
            bf16x8 lo = *(const bf16x8*)(arow1 + k0 + 8 * half);
            bf16x8 hi = *(const bf16x8*)(arow1 + k0 + 16 + 8 * half);
            #pragma unroll
            for (int i = 0; i < 8; i++) { a1[i] = lo[i]; a1[8 + i] = hi[i]; }
        }
        #pragma unroll
        for (int t = 0; t < 4; t++) {
            // B fragment from LDS: K = k0+8*half+0..7 & k0+16+8*half+0..7 for column n0+16t+r
            const __bf16* brow = ldsB + (16 * t + r) * ldsK + k0 + 8 * half;
            bf16x8 b0v = *(const bf16x8*)(brow);
            bf16x8 b1v = *(const bf16x8*)(brow + 16);
            v16bf b;
            #pragma unroll
            for (int i = 0; i < 8; i++) { b[i] = b0v[i]; b[8 + i] = b1v[i]; }
            acc[0][t] = __builtin_amdgcn_wmma_f32_16x16x32_bf16(
                false, a0, false, b, (short)0, acc[0][t], false, false);
            acc[1][t] = __builtin_amdgcn_wmma_f32_16x16x32_bf16(
                false, a1, false, b, (short)0, acc[1][t], false, false);
        }
    }

    #pragma unroll
    for (int mt = 0; mt < 2; mt++) {
        #pragma unroll
        for (int t = 0; t < 4; t++) {
            int n = n0 + 16 * t + r;
            float bv = bias ? bias[n] : 0.0f;
            #pragma unroll
            for (int v = 0; v < 8; v++) {
                int m = m0 + mt * 16 + v + 8 * half;   // C/D layout: VGPR v -> M = v + 8*half
                C[(size_t)m * ldc + n] = acc[mt][t][v] + bv;
            }
        }
    }
}

// ---------------- leaf gates: c0 = sig(i)*tanh(u); h0 = sig(o)*tanh(c0)
__global__ void k_leaf_epi(const float* __restrict__ Ciou,
                           float* __restrict__ h, float* __restrict__ c)
{
    int idx = blockIdx.x * blockDim.x + threadIdx.x;
    if (idx >= NLEAF * TD) return;
    int row = idx / TD, col = idx % TD;
    const float* p = Ciou + (size_t)row * 3 * TD;
    float iv = sigf(p[col]);
    float ov = sigf(p[TD + col]);
    float uv = tanhf(p[2 * TD + col]);
    float c0 = iv * uv;
    float h0 = ov * tanhf(c0);
    c[idx] = c0;
    h[idx] = h0;
}

// ---------------- level A-panel builders ----------------
// Aiou[P x 320] = [ feats[off+j] (192) | bf16(h[c0]+h[c1]) (128) ]
__global__ void k_build_Aiou(const __bf16* __restrict__ feats, const float* __restrict__ h,
                             const int* __restrict__ adj, __bf16* __restrict__ Aiou,
                             int P, int off, int eoff)
{
    int idx = blockIdx.x * blockDim.x + threadIdx.x;
    if (idx >= P * KCAT) return;
    int j = idx / KCAT, col = idx % KCAT;
    if (col < IND) {
        Aiou[idx] = feats[(size_t)(off + j) * IND + col];
    } else {
        int c0 = adj[(size_t)(eoff + 2 * j) * 2 + 1];
        int c1 = adj[(size_t)(eoff + 2 * j + 1) * 2 + 1];
        int cc = col - IND;
        Aiou[idx] = (__bf16)(h[(size_t)c0 * TD + cc] + h[(size_t)c1 * TD + cc]);
    }
}

// Af[2P x 320] = [ feats[parent(m)] (192) | bf16(h[child(m)]) (128) ]
__global__ void k_build_Af(const __bf16* __restrict__ feats, const float* __restrict__ h,
                           const int* __restrict__ adj, __bf16* __restrict__ Af,
                           int P, int eoff)
{
    int idx = blockIdx.x * blockDim.x + threadIdx.x;
    if (idx >= 2 * P * KCAT) return;
    int m = idx / KCAT, col = idx % KCAT;
    const int* e = adj + (size_t)(eoff + m) * 2;
    if (col < IND) Af[idx] = feats[(size_t)e[0] * IND + col];
    else           Af[idx] = (__bf16)h[(size_t)e[1] * TD + (col - IND)];
}

// ---------------- level gates ----------------
__global__ void k_level_epi(const float* __restrict__ Ciou, const float* __restrict__ Cf,
                            const int* __restrict__ adj, float* __restrict__ h,
                            float* __restrict__ c, int P, int off, int eoff)
{
    int idx = blockIdx.x * blockDim.x + threadIdx.x;
    if (idx >= P * TD) return;
    int j = idx / TD, col = idx % TD;
    const float* p = Ciou + (size_t)j * 3 * TD;
    float iv = sigf(p[col]);
    float ov = sigf(p[TD + col]);
    float uv = tanhf(p[2 * TD + col]);
    int c0 = adj[(size_t)(eoff + 2 * j) * 2 + 1];
    int c1 = adj[(size_t)(eoff + 2 * j + 1) * 2 + 1];
    float f0 = sigf(Cf[(size_t)(2 * j) * TD + col]);
    float f1 = sigf(Cf[(size_t)(2 * j + 1) * TD + col]);
    float cs = f0 * c[(size_t)c0 * TD + col] + f1 * c[(size_t)c1 * TD + col];
    float cn = iv * uv + cs;
    float hn = ov * tanhf(cn);
    size_t ridx = (size_t)(off + j) * TD + col;
    c[ridx] = cn;
    h[ridx] = hn;
}

// ---------------- root copy (h[-B:] before h buffer is reused by next tree)
__global__ void k_copy_roots(const float* __restrict__ h, float* __restrict__ roots)
{
    int idx = blockIdx.x * blockDim.x + threadIdx.x;
    if (idx >= BATCH_ * TD) return;
    roots[idx] = h[(size_t)(NNODES - BATCH_) * TD + idx];
}

// ---------------- head: logits = relu([h_c|h_a|h_b|dot] @ W1 + b1) @ W2 + b2
__global__ void k_head(const float* __restrict__ ra, const float* __restrict__ rb,
                       const float* __restrict__ rc,
                       const float* __restrict__ W1, const float* __restrict__ b1,
                       const float* __restrict__ W2, const float* __restrict__ b2,
                       float* __restrict__ out)
{
    __shared__ float hc[3 * TD + 1];
    __shared__ float z1[TD];
    __shared__ float red[TD];
    int row = blockIdx.x, t = threadIdx.x;   // 128 threads
    float av = ra[(size_t)row * TD + t];
    float bv = rb[(size_t)row * TD + t];
    hc[t]          = rc[(size_t)row * TD + t];
    hc[TD + t]     = av;
    hc[2 * TD + t] = bv;
    red[t] = av * bv;
    __syncthreads();
    for (int s = 64; s > 0; s >>= 1) { if (t < s) red[t] += red[t + s]; __syncthreads(); }
    if (t == 0) hc[3 * TD] = red[0];
    __syncthreads();
    float acc = b1[t];
    for (int k = 0; k < 3 * TD + 1; k++) acc += hc[k] * W1[k * TD + t];
    z1[t] = fmaxf(acc, 0.0f);
    __syncthreads();
    if (t < 3) {
        float s = b2[t];
        for (int k = 0; k < TD; k++) s += z1[k] * W2[k * 3 + t];
        out[(size_t)row * 3 + t] = s;
    }
}

// ---------------- driver ----------------
extern "C" void kernel_launch(void* const* d_in, const int* in_sizes, int n_in,
                              void* d_out, int out_size, void* d_ws, size_t ws_size,
                              hipStream_t stream)
{
    // input order: tok_c,srt_c,const_c, tok_a,srt_a,const_a, tok_b,srt_b,const_b,
    //              adjacency, D, B, emb_table, sort_table, W_iou,b_iou,U_iou,
    //              W_f,b_f,U_f, W1,b1,W2,b2
    const int*   tok[3] = { (const int*)d_in[3], (const int*)d_in[6], (const int*)d_in[0] }; // a,b,c
    const int*   srt[3] = { (const int*)d_in[4], (const int*)d_in[7], (const int*)d_in[1] };
    const float* cst[3] = { (const float*)d_in[5], (const float*)d_in[8], (const float*)d_in[2] };
    const int*   adj    = (const int*)d_in[9];
    const float* emb    = (const float*)d_in[12];
    const float* sortt  = (const float*)d_in[13];
    const float* W_iou  = (const float*)d_in[14];
    const float* b_iou  = (const float*)d_in[15];
    const float* U_iou  = (const float*)d_in[16];
    const float* W_f    = (const float*)d_in[17];
    const float* b_f    = (const float*)d_in[18];
    const float* U_f    = (const float*)d_in[19];
    const float* W1     = (const float*)d_in[20];
    const float* b1     = (const float*)d_in[21];
    const float* W2     = (const float*)d_in[22];
    const float* b2     = (const float*)d_in[23];

    char* ws = (char*)d_ws;
    size_t off_ = 0;
    auto alloc = [&](size_t bytes) { size_t r = off_; off_ += (bytes + 255) & ~(size_t)255; return r; };

    __bf16* WiouT = (__bf16*)(ws + alloc((size_t)384 * KCAT * 2));
    __bf16* WfT   = (__bf16*)(ws + alloc((size_t)TD * KCAT * 2));
    __bf16* feats = (__bf16*)(ws + alloc((size_t)NNODES * IND * 2));
    float*  hbuf  = (float*)(ws + alloc((size_t)NNODES * TD * 4));
    float*  cbuf  = (float*)(ws + alloc((size_t)NNODES * TD * 4));
    __bf16* Aiou  = (__bf16*)(ws + alloc((size_t)32768 * KCAT * 2));
    __bf16* Af    = (__bf16*)(ws + alloc((size_t)65536 * KCAT * 2));
    float*  Ciou  = (float*)(ws + alloc((size_t)NLEAF * 3 * TD * 4));   // reused by levels
    float*  Cf    = (float*)(ws + alloc((size_t)65536 * TD * 4));
    float*  roots = (float*)(ws + alloc((size_t)3 * BATCH_ * TD * 4));

    // weights -> stacked transposed bf16
    {
        int tot = 384 * KCAT;
        k_build_wt<<<(tot + 255) / 256, 256, 0, stream>>>(W_iou, U_iou, WiouT, 384, IND, TD);
        tot = TD * KCAT;
        k_build_wt<<<(tot + 255) / 256, 256, 0, stream>>>(W_f, U_f, WfT, TD, IND, TD);
    }

    for (int t = 0; t < 3; t++) {
        int tot = NNODES * IND;
        k_feats<<<(tot + 255) / 256, 256, 0, stream>>>(tok[t], srt[t], cst[t], emb, sortt, feats);

        // leaves: [65536 x 192] @ [192 x 384] (Bt = first 192 K-columns of WiouT, ldb = 320)
        {
            dim3 g(384 / 64, NLEAF / 256);
            k_gemm<<<g, 256, 0, stream>>>(feats, IND, WiouT, KCAT, b_iou, Ciou, 3 * TD, IND);
            tot = NLEAF * TD;
            k_leaf_epi<<<(tot + 255) / 256, 256, 0, stream>>>(Ciou, hbuf, cbuf);
        }

        int nodeOff = NLEAF, eoff = 0;
        for (int n = 1; n <= DEPTH_; n++) {
            int P = BATCH_ << (DEPTH_ - n);
            tot = P * KCAT;
            k_build_Aiou<<<(tot + 255) / 256, 256, 0, stream>>>(feats, hbuf, adj, Aiou, P, nodeOff, eoff);
            tot = 2 * P * KCAT;
            k_build_Af<<<(tot + 255) / 256, 256, 0, stream>>>(feats, hbuf, adj, Af, P, eoff);

            dim3 g1(384 / 64, P / 256);
            k_gemm<<<g1, 256, 0, stream>>>(Aiou, KCAT, WiouT, KCAT, b_iou, Ciou, 3 * TD, KCAT);
            dim3 g2(TD / 64, (2 * P) / 256);
            k_gemm<<<g2, 256, 0, stream>>>(Af, KCAT, WfT, KCAT, b_f, Cf, TD, KCAT);

            tot = P * TD;
            k_level_epi<<<(tot + 255) / 256, 256, 0, stream>>>(Ciou, Cf, adj, hbuf, cbuf, P, nodeOff, eoff);
            eoff += 2 * P;
            nodeOff += P;
        }
        tot = BATCH_ * TD;
        k_copy_roots<<<(tot + 255) / 256, 256, 0, stream>>>(hbuf, roots + (size_t)t * BATCH_ * TD);
    }

    // roots: slot 0 = tree a, 1 = tree b, 2 = tree c
    k_head<<<BATCH_, TD, 0, stream>>>(roots, roots + (size_t)BATCH_ * TD,
                                      roots + (size_t)2 * BATCH_ * TD,
                                      W1, b1, W2, b2, (float*)d_out);
}
